// NeuralODE_56238301774293
// MI455X (gfx1250) — matrix operand order repaired and checked by reading, not compiled, not run on hardware
//
#include <hip/hip_runtime.h>

#define B_ROWS 8192
#define DIM    512
#define NTILES (DIM / 16)     // 32 column tiles
#define KCHUNKS (DIM / 32)    // 16 k-chunks of 32 (bf16 wmma K)
#define STEPS  20
#define WAVES_PER_WG 4
#define ROWS_PER_WAVE 16
#define ROWS_PER_WG (WAVES_PER_WG * ROWS_PER_WAVE)   // 64
#define LDSS 520              // padded LDS row stride (elements) -> conflict-reduced

typedef __bf16 v16bf __attribute__((ext_vector_type(16)));
typedef float  v8f   __attribute__((ext_vector_type(8)));

union ABReg { v16bf v; uint4 q[2]; };

__device__ __forceinline__ unsigned short f2bf(float f) {
  // round-to-nearest-even fp32 -> bf16
  unsigned int u = __float_as_uint(f);
  u += 0x7FFFu + ((u >> 16) & 1u);
  return (unsigned short)(u >> 16);
}

__device__ __forceinline__ float fast_tanh(float x) {
#if __has_builtin(__builtin_amdgcn_tanhf)
  return __builtin_amdgcn_tanhf(x);          // v_tanh_f32 (gfx1250 TRANS op)
#elif __has_builtin(__builtin_amdgcn_exp2f) && __has_builtin(__builtin_amdgcn_rcpf)
  // branch-free: tanh(x) = 1 - 2/(e^{2x}+1); exp2 overflow/underflow saturate to +/-1
  float e = __builtin_amdgcn_exp2f(x * 2.885390081777927f);  // e^{2x}
  float r = __builtin_amdgcn_rcpf(e + 1.0f);
  return __builtin_fmaf(-2.0f, r, 1.0f);
#else
  return tanhf(x);
#endif
}

// One-time prep: transpose+convert W1,W2 to bf16 [n][k], copy z0 -> Z (d_out).
__global__ void prep_kernel(const float* __restrict__ W1,
                            const float* __restrict__ W2,
                            const float* __restrict__ z0,
                            unsigned short* __restrict__ W1t,
                            unsigned short* __restrict__ W2t,
                            float* __restrict__ Z) {
  int idx  = blockIdx.x * blockDim.x + threadIdx.x;
  int nthr = gridDim.x * blockDim.x;
  if (idx < DIM * DIM) {
    int n = idx >> 9;          // idx / 512
    int k = idx & (DIM - 1);   // idx % 512
    W1t[idx] = f2bf(W1[k * DIM + n]);
    W2t[idx] = f2bf(W2[k * DIM + n]);
  }
  for (int i = idx; i < B_ROWS * DIM; i += nthr) Z[i] = z0[i];
}

__global__ __launch_bounds__(WAVES_PER_WG * 32, 1)
void ode_kernel(const unsigned short* __restrict__ W1t,
                const unsigned short* __restrict__ W2t,
                const float* __restrict__ b1,
                const float* __restrict__ b2,
                const float* __restrict__ t,
                float* __restrict__ Z) {
  __shared__ __align__(16) unsigned short Zb[WAVES_PER_WG][ROWS_PER_WAVE][LDSS];
  __shared__ __align__(16) unsigned short Hb[WAVES_PER_WG][ROWS_PER_WAVE][LDSS];

  const int lane = threadIdx.x & 31;
  const int wave = threadIdx.x >> 5;
  const int l15  = lane & 15;
  const int half = lane >> 4;
  const int row0 = blockIdx.x * ROWS_PER_WG + wave * ROWS_PER_WAVE;

  const float h = (t[1] - t[0]) * (1.0f / (float)STEPS);

  // Load this wave's 16 fp32 rows of z (from d_out, written by prep) into bf16 LDS.
  for (int r = 0; r < ROWS_PER_WAVE; ++r) {
    for (int c = lane; c < DIM; c += 32) {
      Zb[wave][r][c] = f2bf(Z[(row0 + r) * DIM + c]);
    }
  }
  // No barrier: all state below is wave-private.

  for (int step = 0; step < STEPS; ++step) {
    // ---------- layer 1: H = tanh(Z @ W1 + b1) ----------
    for (int nt = 0; nt < NTILES; ++nt) {
      const float bias = b1[nt * 16 + l15];
      v8f acc;
      #pragma unroll
      for (int i = 0; i < 8; ++i) acc[i] = bias;

      const unsigned short* __restrict__ wrow = W1t + (size_t)(nt * 16 + l15) * DIM;
      #pragma unroll 8
      for (int kc = 0; kc < KCHUNKS; ++kc) {
        const int kb = kc * 32;
        ABReg a, b;
        // A (16-bit 16x32 layout): VGPR0-3 = K[kb + half*8 ..], VGPR4-7 = +16
        a.q[0] = *(const uint4*)&Zb[wave][l15][kb + half * 8];
        a.q[1] = *(const uint4*)&Zb[wave][l15][kb + 16 + half * 8];
        // B (32x16): same K order as A: VGPR0-3 = K[kb + half*8 ..], VGPR4-7 = +16, from Wt[n][k]
        b.q[0] = *(const uint4*)&wrow[kb + half * 8];
        b.q[1] = *(const uint4*)&wrow[kb + 16 + half * 8];
        acc = __builtin_amdgcn_wmma_f32_16x16x32_bf16(
            false, a.v, false, b.v, (short)0, acc, false, false);
      }
      // C/D layout: VGPR r -> row (r + half*8), col (nt*16 + l15)
      #pragma unroll
      for (int r = 0; r < 8; ++r) {
        Hb[wave][r + half * 8][nt * 16 + l15] = f2bf(fast_tanh(acc[r]));
      }
    }

    // ---------- layer 2: Z += h * (H @ W2 + b2) ----------
    for (int nt = 0; nt < NTILES; ++nt) {
      const float bias = b2[nt * 16 + l15];
      v8f acc;
      #pragma unroll
      for (int i = 0; i < 8; ++i) acc[i] = bias;

      const unsigned short* __restrict__ wrow = W2t + (size_t)(nt * 16 + l15) * DIM;
      #pragma unroll 8
      for (int kc = 0; kc < KCHUNKS; ++kc) {
        const int kb = kc * 32;
        ABReg a, b;
        a.q[0] = *(const uint4*)&Hb[wave][l15][kb + half * 8];
        a.q[1] = *(const uint4*)&Hb[wave][l15][kb + 16 + half * 8];
        b.q[0] = *(const uint4*)&wrow[kb + half * 8];
        b.q[1] = *(const uint4*)&wrow[kb + 16 + half * 8];
        acc = __builtin_amdgcn_wmma_f32_16x16x32_bf16(
            false, a.v, false, b.v, (short)0, acc, false, false);
      }
      #pragma unroll
      for (int r = 0; r < 8; ++r) {
        const int gr = row0 + r + half * 8;
        const int gc = nt * 16 + l15;
        const float zn = Z[(size_t)gr * DIM + gc] + h * acc[r];
        Z[(size_t)gr * DIM + gc] = zn;          // fp32 master state (wave-private rows)
        Zb[wave][r + half * 8][gc] = f2bf(zn);  // bf16 A operand for next step
      }
    }
  }
  // Final z already resides in Z == d_out.
}

extern "C" void kernel_launch(void* const* d_in, const int* in_sizes, int n_in,
                              void* d_out, int out_size, void* d_ws, size_t ws_size,
                              hipStream_t stream) {
  const float* z0 = (const float*)d_in[0];
  const float* t  = (const float*)d_in[1];
  const float* W1 = (const float*)d_in[2];
  const float* b1 = (const float*)d_in[3];
  const float* W2 = (const float*)d_in[4];
  const float* b2 = (const float*)d_in[5];
  float* Z = (float*)d_out;

  unsigned short* W1t = (unsigned short*)d_ws;        // 512*512 bf16 = 512 KB
  unsigned short* W2t = W1t + DIM * DIM;              // next 512 KB

  prep_kernel<<<1024, 256, 0, stream>>>(W1, W2, z0, W1t, W2t, Z);
  ode_kernel<<<B_ROWS / ROWS_PER_WG, WAVES_PER_WG * 32, 0, stream>>>(
      W1t, W2t, b1, b2, t, Z);
}
